// JetFFNMoEBlock_7937099563425
// MI455X (gfx1250) — matrix operand order repaired and checked by reading, not compiled, not run on hardware
//
#include <hip/hip_runtime.h>
#include <math.h>

// ---------------------------------------------------------------------------
// JetFFNMoEBlock for MI455X (gfx1250): bf16 WMMA + async global->LDS staging.
//   B=4, S=1024, H=1024 (16 heads x 64), F=4096, E=8 experts, top-2.
// ---------------------------------------------------------------------------

#define H_DIM   1024
#define F_DIM   4096
#define NHEAD   16
#define HEAD_DIM 64
#define NEXP    8
#define BATCH   4
#define SEQ     1024
#define NTOK    4096   // BATCH*SEQ

typedef __attribute__((ext_vector_type(16))) __bf16 v16bf;
typedef __attribute__((ext_vector_type(8)))  float  v8f;

union Frag {
  v16bf v;
  uint4 q[2];
};

__device__ __forceinline__ v8f wmma_bf16(const Frag& a, const Frag& b, v8f c) {
  // D = A(16x32 bf16) * B(32x16 bf16) + C(16x16 f32)
  return __builtin_amdgcn_wmma_f32_16x16x32_bf16(
      /*neg_a=*/false, a.v, /*neg_b=*/false, b.v,
      /*c_mod=*/(short)0, c, /*reuse_a=*/false, /*reuse_b=*/false);
}

// CDNA5 async global->LDS copy (ASYNCcnt tracked). Per lane: 16 bytes from the
// lane's 64-bit global address into the lane's 32-bit LDS address.
__device__ __forceinline__ void async_ld_b128(unsigned lds_off, const void* gptr) {
  asm volatile("global_load_async_to_lds_b128 %0, %1, off"
               :: "v"(lds_off), "v"(gptr)
               : "memory");
}
__device__ __forceinline__ void wait_async_le4() {
  asm volatile("s_wait_asynccnt 0x4" ::: "memory");
}
__device__ __forceinline__ void wait_async_0() {
  asm volatile("s_wait_asynccnt 0x0" ::: "memory");
}

// f32 <-> bf16 (bit-level, round-to-nearest-even)
__device__ __forceinline__ unsigned short f2b(float f) {
  union { float f; unsigned u; } a; a.f = f;
  unsigned r = a.u + 0x7FFFu + ((a.u >> 16) & 1u);
  return (unsigned short)(r >> 16);
}
__device__ __forceinline__ float b2f(unsigned short h) {
  union { unsigned u; float f; } a; a.u = ((unsigned)h) << 16;
  return a.f;
}
__device__ __forceinline__ float gelu_f(float x) {
  const float k = 0.7978845608028654f; // sqrt(2/pi)
  float x3 = x * x * x;
  return 0.5f * x * (1.0f + tanhf(k * (x + 0.044715f * x3)));
}

// ---------------------------------------------------------------------------
// Weight transpose + f32->bf16:  Wt[c*R + r] = W[r*C + c]   (Bt[N][K] layout)
// ---------------------------------------------------------------------------
__global__ __launch_bounds__(256) void transpose_f32_to_bf16(
    const float* __restrict__ W, unsigned short* __restrict__ Wt, int R, int C) {
  size_t i = (size_t)blockIdx.x * 256 + threadIdx.x;
  if (i >= (size_t)R * C) return;
  int c = (int)(i / R);
  int r = (int)(i % R);
  Wt[i] = f2b(W[(size_t)r * C + c]);
}

// ---------------------------------------------------------------------------
// RMSNorm: one token (row of 1024 f32) per block of 256; bf16 output
// ---------------------------------------------------------------------------
__global__ __launch_bounds__(256) void rmsnorm_kernel(
    const float* __restrict__ x, const float* __restrict__ g,
    unsigned short* __restrict__ out) {
  int t = blockIdx.x;
  const float* row = x + (size_t)t * H_DIM;
  int tid = threadIdx.x;
  float vals[4];
  float ss = 0.f;
  for (int i = 0; i < 4; ++i) { vals[i] = row[tid + i * 256]; ss += vals[i] * vals[i]; }
  for (int off = 16; off; off >>= 1) ss += __shfl_xor(ss, off, 32);
  __shared__ float red[8];
  int wave = tid >> 5, lane = tid & 31;
  if (lane == 0) red[wave] = ss;
  __syncthreads();
  if (tid == 0) {
    float tot = 0.f;
    for (int w = 0; w < 8; ++w) tot += red[w];
    red[0] = rsqrtf(tot / (float)H_DIM + 1e-5f);
  }
  __syncthreads();
  float rs = red[0];
  for (int i = 0; i < 4; ++i) {
    int c = tid + i * 256;
    out[(size_t)t * H_DIM + c] = f2b(vals[i] * rs * g[c]);
  }
}

// ---------------------------------------------------------------------------
// Generic 128x128 bf16 WMMA GEMM: C[M,N] = A[M,K] * Bt[N,K]^T
//   8 waves, each a 32x64 slab as 2x4 v_wmma_f32_16x16x32_bf16 accums.
//   K-tiles of 32 double-buffered in LDS via global_load_async_to_lds_b128:
//   each wave issues 4 async b128s/stage; loads retire in order, so
//   s_wait_asynccnt<=4 after issuing the next stage means the current stage
//   is resident; a workgroup barrier then publishes it to all 8 waves.
//   modes: 0 = store f32; 1 = +resid, store f32; 2 = gelu -> bf16 store;
//          3 = atomicAdd(C[rowmapOut[m]], gate[m]*acc)  (MoE scatter)
// ---------------------------------------------------------------------------
__global__ __launch_bounds__(256) void gemm_bf16_kernel(
    const unsigned short* __restrict__ A, int lda,
    const unsigned short* __restrict__ Bt, int ldb,
    int K,
    float* __restrict__ Cf, int ldc,
    const float* __restrict__ resid,
    unsigned short* __restrict__ Cbf,
    const int* __restrict__ rowmapA,
    const int* __restrict__ rowmapOut,
    const float* __restrict__ gates,
    const int* __restrict__ cntPtr,
    int mode) {
  __shared__ __align__(16) unsigned short sA[2][128][40];  // 32 + 8 pad per row
  __shared__ __align__(16) unsigned short sB[2][128][40];

  const int m0 = blockIdx.y * 128;
  const int n0 = blockIdx.x * 128;
  if (cntPtr && m0 >= *cntPtr) return;  // uniform early exit for MoE tiles

  const int tid = threadIdx.x;
  const int wave = tid >> 5, lane = tid & 31;
  const int hh = lane >> 4, lm = lane & 15;
  const int wm = (wave >> 1) * 32;  // 4 wave rows -> 128 M
  const int wn = (wave & 1) * 64;   // 2 wave cols -> 128 N

  // cooperative staging: 2 x uint4 (8 bf16) per thread per matrix per stage
  int ar[2], ac[2];
  const unsigned short* aPtr[2];
  const unsigned short* bPtr[2];
  unsigned ldsA[2][2], ldsB[2][2];  // [buffer][i]
  for (int i = 0; i < 2; ++i) {
    int u = tid + i * 256;
    ar[i] = u >> 2;
    ac[i] = (u & 3) * 8;
    int grow = m0 + ar[i];
    int arow = rowmapA ? rowmapA[grow] : grow;
    aPtr[i] = A + (size_t)arow * lda + ac[i];
    bPtr[i] = Bt + (size_t)(n0 + ar[i]) * ldb + ac[i];
    for (int buf = 0; buf < 2; ++buf) {
      ldsA[buf][i] = (unsigned)(uintptr_t)&sA[buf][ar[i]][ac[i]];
      ldsB[buf][i] = (unsigned)(uintptr_t)&sB[buf][ar[i]][ac[i]];
    }
  }

  v8f acc[2][4];
  for (int i = 0; i < 2; ++i)
    for (int j = 0; j < 4; ++j)
      for (int r = 0; r < 8; ++r) acc[i][j][r] = 0.0f;

  // prologue: stage k=0 into buffer 0
  for (int i = 0; i < 2; ++i) {
    async_ld_b128(ldsA[0][i], aPtr[i]);
    async_ld_b128(ldsB[0][i], bPtr[i]);
  }

  for (int k0 = 0; k0 < K; k0 += 32) {
    const int cur = (k0 >> 5) & 1;
    const int nxt = cur ^ 1;
    if (k0 + 32 < K) {
      for (int i = 0; i < 2; ++i) {
        async_ld_b128(ldsA[nxt][i], aPtr[i] + k0 + 32);
        async_ld_b128(ldsB[nxt][i], bPtr[i] + k0 + 32);
      }
      wait_async_le4();   // current stage's 4 loads have landed in LDS
    } else {
      wait_async_0();
    }
    __syncthreads();

    // A fragment (ISA 16-bit A layout): lane m=lm, half hh: K = {8hh..+7, 16+8hh..+7}
    Frag a[2], b[4];
    for (int i = 0; i < 2; ++i) {
      int m = wm + i * 16 + lm;
      a[i].q[0] = *(const uint4*)&sA[cur][m][8 * hh];
      a[i].q[1] = *(const uint4*)&sA[cur][m][16 + 8 * hh];
    }
    // B fragment (Bt rows are N): lane n=lm, half hh: K = {8hh..+7, 16+8hh..+7}
    for (int j = 0; j < 4; ++j) {
      int n = wn + j * 16 + lm;
      b[j].q[0] = *(const uint4*)&sB[cur][n][8 * hh];
      b[j].q[1] = *(const uint4*)&sB[cur][n][16 + 8 * hh];
    }
    for (int i = 0; i < 2; ++i)
      for (int j = 0; j < 4; ++j)
        acc[i][j] = wmma_bf16(a[i], b[j], acc[i][j]);
    __syncthreads();  // all waves done with 'cur' before it is refilled
  }

  // epilogue (C layout: M = r + 8*hh, N = lm)
  for (int i = 0; i < 2; ++i) {
    for (int j = 0; j < 4; ++j) {
      for (int r = 0; r < 8; ++r) {
        int gm = m0 + wm + i * 16 + r + 8 * hh;
        int gn = n0 + wn + j * 16 + lm;
        float val = acc[i][j][r];
        if (mode == 0) {
          Cf[(size_t)gm * ldc + gn] = val;
        } else if (mode == 1) {
          Cf[(size_t)gm * ldc + gn] = val + resid[(size_t)gm * ldc + gn];
        } else if (mode == 2) {
          Cbf[(size_t)gm * ldc + gn] = f2b(gelu_f(val));
        } else {
          atomicAdd(&Cf[(size_t)rowmapOut[gm] * ldc + gn], gates[gm] * val);
        }
      }
    }
  }
}

// ---------------------------------------------------------------------------
// RoPE + layout reorg: qkv f32 [T,3072] -> q,k bf16 [b,h,s,64], v^T bf16 [b,h,64,s]
// ---------------------------------------------------------------------------
__global__ __launch_bounds__(256) void rope_reorg_kernel(
    const float* __restrict__ qkv,
    unsigned short* __restrict__ qro,
    unsigned short* __restrict__ kro,
    unsigned short* __restrict__ vtr) {
  int gid = blockIdx.x * 256 + threadIdx.x;  // B*S*NH*32
  int d = gid & 31;
  int head = (gid >> 5) & 15;
  int s = (gid >> 9) & 1023;
  int b = gid >> 19;
  const float* row = qkv + (size_t)(b * SEQ + s) * 3072;
  float inv = __powf(10000.f, -(float)d * (1.0f / 32.0f));
  float sn, cs;
  __sincosf((float)s * inv, &sn, &cs);
  int ho = head * 64;
  float q1 = row[ho + d], q2 = row[ho + d + 32];
  float k1 = row[1024 + ho + d], k2 = row[1024 + ho + d + 32];
  size_t ob = (((size_t)(b * NHEAD + head)) * SEQ + s) * 64;
  qro[ob + d]      = f2b(q1 * cs - q2 * sn);
  qro[ob + d + 32] = f2b(q2 * cs + q1 * sn);
  kro[ob + d]      = f2b(k1 * cs - k2 * sn);
  kro[ob + d + 32] = f2b(k2 * cs + k1 * sn);
  size_t vb = ((size_t)(b * NHEAD + head)) * 64 * SEQ;
  vtr[vb + (size_t)d * SEQ + s]        = f2b(row[2048 + ho + d]);
  vtr[vb + (size_t)(d + 32) * SEQ + s] = f2b(row[2048 + ho + d + 32]);
}

// ---------------------------------------------------------------------------
// Causal flash attention: each wave owns 16 query rows, 32-key chunks.
//   S = Q K^T (4 wmma), online softmax (shfl width-16 row reduce),
//   P restaged through per-wave LDS into A layout, O += P V (4 wmma).
// ---------------------------------------------------------------------------
__global__ __launch_bounds__(256) void flash_attn_kernel(
    const unsigned short* __restrict__ qro,
    const unsigned short* __restrict__ kro,
    const unsigned short* __restrict__ vtr,
    unsigned short* __restrict__ attnb) {
  __shared__ __align__(16) unsigned short sP[8][16][40];
  int tid = threadIdx.x;
  int wave = tid >> 5, lane = tid & 31;
  int hh = lane >> 4, lm = lane & 15;
  int q0 = blockIdx.x * 128 + wave * 16;
  int head = blockIdx.y, b = blockIdx.z;
  const unsigned short* Q  = qro + (((size_t)(b * NHEAD + head)) * SEQ + q0) * 64;
  const unsigned short* Kb = kro + ((size_t)(b * NHEAD + head)) * SEQ * 64;
  const unsigned short* Vt = vtr + ((size_t)(b * NHEAD + head)) * 64 * SEQ;

  Frag qa[2];
  {
    const unsigned short* qrow = Q + lm * 64;
    for (int c = 0; c < 2; ++c) {
      qa[c].q[0] = *(const uint4*)(qrow + c * 32 + 8 * hh);
      qa[c].q[1] = *(const uint4*)(qrow + c * 32 + 16 + 8 * hh);
    }
  }

  float mrow[8], lrow[8];
  v8f o[4];
  for (int r = 0; r < 8; ++r) { mrow[r] = -1e30f; lrow[r] = 0.f; }
  for (int j = 0; j < 4; ++j)
    for (int r = 0; r < 8; ++r) o[j][r] = 0.f;

  int kend = q0 + 16;  // exclusive upper bound of needed keys
  for (int kc = 0; kc < kend; kc += 32) {
    v8f sacc[2];
    for (int st = 0; st < 2; ++st)
      for (int r = 0; r < 8; ++r) sacc[st][r] = 0.f;
    for (int st = 0; st < 2; ++st) {
      const unsigned short* krow = Kb + (size_t)(kc + st * 16 + lm) * 64;
      for (int c = 0; c < 2; ++c) {
        Frag kb;
        kb.q[0] = *(const uint4*)(krow + c * 32 + 8 * hh);
        kb.q[1] = *(const uint4*)(krow + c * 32 + 16 + 8 * hh);
        sacc[st] = wmma_bf16(qa[c], kb, sacc[st]);
      }
    }
    for (int r = 0; r < 8; ++r) {
      int rowq = q0 + r + 8 * hh;
      float s0 = sacc[0][r] * 0.125f;  // 1/sqrt(64)
      float s1 = sacc[1][r] * 0.125f;
      if (kc + lm > rowq)      s0 = -1e9f;
      if (kc + 16 + lm > rowq) s1 = -1e9f;
      float mx = fmaxf(s0, s1);
      for (int off = 8; off; off >>= 1) mx = fmaxf(mx, __shfl_xor(mx, off, 16));
      float mn = fmaxf(mrow[r], mx);
      float p0 = __expf(s0 - mn);
      float p1 = __expf(s1 - mn);
      float corr = __expf(mrow[r] - mn);
      float rs = p0 + p1;
      for (int off = 8; off; off >>= 1) rs += __shfl_xor(rs, off, 16);
      lrow[r] = lrow[r] * corr + rs;
      mrow[r] = mn;
      for (int j = 0; j < 4; ++j) o[j][r] *= corr;
      int prow = r + 8 * hh;
      sP[wave][prow][lm]      = f2b(p0);
      sP[wave][prow][16 + lm] = f2b(p1);
    }
    // re-read P in A-fragment layout (wave-private LDS; DS counters order this)
    Frag pa;
    pa.q[0] = *(const uint4*)&sP[wave][lm][8 * hh];
    pa.q[1] = *(const uint4*)&sP[wave][lm][16 + 8 * hh];
    for (int j = 0; j < 4; ++j) {
      Frag vbf;
      const unsigned short* vrow = Vt + (size_t)(j * 16 + lm) * SEQ + kc + 8 * hh;
      vbf.q[0] = *(const uint4*)vrow;
      vbf.q[1] = *(const uint4*)(vrow + 16);
      o[j] = wmma_bf16(pa, vbf, o[j]);
    }
  }
  for (int j = 0; j < 4; ++j) {
    for (int r = 0; r < 8; ++r) {
      int rowq = q0 + r + 8 * hh;
      float val = o[j][r] / lrow[r];
      attnb[((size_t)(b * SEQ + rowq)) * H_DIM + head * 64 + j * 16 + lm] = f2b(val);
    }
  }
}

// ---------------------------------------------------------------------------
// Router: one wave per token. logits = xn2 . w_router, softmax, top-2,
// gather lists via atomics, accumulate prob sums for aux loss.
// ---------------------------------------------------------------------------
__global__ __launch_bounds__(256) void router_kernel(
    const unsigned short* __restrict__ xn2, const float* __restrict__ wr,
    int* __restrict__ etok, float* __restrict__ egate,
    int* __restrict__ cnt, float* __restrict__ probsum) {
  int t = blockIdx.x * 8 + (threadIdx.x >> 5);
  int lane = threadIdx.x & 31;
  const unsigned short* row = xn2 + (size_t)t * H_DIM;
  float acc[8];
  for (int e = 0; e < 8; ++e) acc[e] = 0.f;
  for (int d = lane; d < H_DIM; d += 32) {
    float xv = b2f(row[d]);
    const float* w = wr + (size_t)d * 8;
    for (int e = 0; e < 8; ++e) acc[e] += xv * w[e];
  }
  for (int e = 0; e < 8; ++e)
    for (int off = 16; off; off >>= 1) acc[e] += __shfl_xor(acc[e], off, 32);
  if (lane == 0) {
    float mx = acc[0];
    for (int e = 1; e < 8; ++e) mx = fmaxf(mx, acc[e]);
    float p[8], sum = 0.f;
    for (int e = 0; e < 8; ++e) { p[e] = __expf(acc[e] - mx); sum += p[e]; }
    float invs = 1.f / sum;
    int i0 = 0;
    for (int e = 0; e < 8; ++e) {
      p[e] *= invs;
      atomicAdd(&probsum[e], p[e]);
      if (p[e] > p[i0]) i0 = e;
    }
    int i1 = (i0 == 0) ? 1 : 0;
    for (int e = 0; e < 8; ++e)
      if (e != i0 && p[e] > p[i1]) i1 = e;
    float denom = p[i0] + p[i1];
    float w0 = p[i0] / denom, w1 = p[i1] / denom;
    int s0 = atomicAdd(&cnt[i0], 1);
    etok[i0 * NTOK + s0] = t; egate[i0 * NTOK + s0] = w0;
    int s1 = atomicAdd(&cnt[i1], 1);
    etok[i1 * NTOK + s1] = t; egate[i1 * NTOK + s1] = w1;
  }
}

__global__ void aux_kernel(const int* __restrict__ cnt,
                           const float* __restrict__ probsum,
                           float* __restrict__ outp) {
  if (threadIdx.x == 0 && blockIdx.x == 0) {
    float a = 0.f;
    for (int e = 0; e < 8; ++e)
      a += ((float)cnt[e] / (float)(NTOK * 2)) * (probsum[e] / (float)NTOK);
    *outp = 8.f * a;  // E * sum(f*P)
  }
}

__global__ __launch_bounds__(256) void add_kernel(const float* __restrict__ a,
                                                  const float* __restrict__ b,
                                                  float* __restrict__ o) {
  size_t i = ((size_t)blockIdx.x * 256 + threadIdx.x) * 4;
  for (int k = 0; k < 4; ++k) o[i + k] = a[i + k] + b[i + k];
}

// ---------------------------------------------------------------------------
// Host orchestration
// ---------------------------------------------------------------------------
extern "C" void kernel_launch(void* const* d_in, const int* in_sizes, int n_in,
                              void* d_out, int out_size, void* d_ws, size_t ws_size,
                              hipStream_t stream) {
  (void)in_sizes; (void)n_in; (void)ws_size;
  const float* x        = (const float*)d_in[0];
  const float* g_attn   = (const float*)d_in[1];
  const float* wq       = (const float*)d_in[2];
  const float* wk       = (const float*)d_in[3];
  const float* wv       = (const float*)d_in[4];
  const float* wo       = (const float*)d_in[5];
  const float* g_ffn    = (const float*)d_in[6];
  const float* w_router = (const float*)d_in[7];
  const float* w1       = (const float*)d_in[8];
  const float* w2       = (const float*)d_in[9];
  float* out = (float*)d_out;

  char* ws = (char*)d_ws;
  size_t off = 0;
  auto alloc = [&](size_t bytes) -> char* {
    char* p = ws + off;
    off = (off + bytes + 255) & ~(size_t)255;
    return p;
  };
  unsigned short* wqkvT  = (unsigned short*)alloc((size_t)3 * H_DIM * H_DIM * 2);
  unsigned short* woT    = (unsigned short*)alloc((size_t)H_DIM * H_DIM * 2);
  unsigned short* w1t    = (unsigned short*)alloc((size_t)NEXP * F_DIM * H_DIM * 2);
  unsigned short* w2t    = (unsigned short*)alloc((size_t)NEXP * H_DIM * F_DIM * 2);
  unsigned short* xn     = (unsigned short*)alloc((size_t)NTOK * H_DIM * 2);
  float*          qkv    = (float*)alloc((size_t)NTOK * 3072 * 4);
  unsigned short* qro    = (unsigned short*)alloc((size_t)NTOK * H_DIM * 2);
  unsigned short* kro    = (unsigned short*)alloc((size_t)NTOK * H_DIM * 2);
  unsigned short* vtr    = (unsigned short*)alloc((size_t)NTOK * H_DIM * 2);
  unsigned short* attnb  = (unsigned short*)alloc((size_t)NTOK * H_DIM * 2);
  float*          x2     = (float*)alloc((size_t)NTOK * H_DIM * 4);
  unsigned short* xn2    = (unsigned short*)alloc((size_t)NTOK * H_DIM * 2);
  float*          moe    = (float*)alloc((size_t)NTOK * H_DIM * 4);
  unsigned short* hidden = (unsigned short*)alloc((size_t)NTOK * F_DIM * 2);
  int*            etok   = (int*)alloc((size_t)NEXP * NTOK * 4);
  float*          egate  = (float*)alloc((size_t)NEXP * NTOK * 4);
  int*            cnt    = (int*)alloc(64);
  float*          probsum= (float*)alloc(64);

  // 1) weight conversion to transposed bf16 (Bt[N][K] layout)
  int tg1 = (H_DIM * H_DIM) / 256;
  transpose_f32_to_bf16<<<tg1, 256, 0, stream>>>(wq, wqkvT,                 H_DIM, H_DIM);
  transpose_f32_to_bf16<<<tg1, 256, 0, stream>>>(wk, wqkvT + H_DIM * H_DIM, H_DIM, H_DIM);
  transpose_f32_to_bf16<<<tg1, 256, 0, stream>>>(wv, wqkvT + 2 * H_DIM * H_DIM, H_DIM, H_DIM);
  transpose_f32_to_bf16<<<tg1, 256, 0, stream>>>(wo, woT, H_DIM, H_DIM);
  int tg2 = (H_DIM * F_DIM) / 256;
  for (int e = 0; e < NEXP; ++e) {
    transpose_f32_to_bf16<<<tg2, 256, 0, stream>>>(
        w1 + (size_t)e * H_DIM * F_DIM, w1t + (size_t)e * F_DIM * H_DIM, H_DIM, F_DIM);
    transpose_f32_to_bf16<<<tg2, 256, 0, stream>>>(
        w2 + (size_t)e * F_DIM * H_DIM, w2t + (size_t)e * H_DIM * F_DIM, F_DIM, H_DIM);
  }

  // 2) zero scratch used with atomics / gather defaults
  hipMemsetAsync(moe,   0, (size_t)NTOK * H_DIM * 4, stream);
  hipMemsetAsync(etok,  0, (size_t)NEXP * NTOK * 4, stream);
  hipMemsetAsync(egate, 0, (size_t)NEXP * NTOK * 4, stream);
  hipMemsetAsync(cnt,   0, 64, stream);
  hipMemsetAsync(probsum, 0, 64, stream);

  // 3) attention path
  rmsnorm_kernel<<<NTOK, 256, 0, stream>>>(x, g_attn, xn);
  gemm_bf16_kernel<<<dim3(3072 / 128, NTOK / 128), 256, 0, stream>>>(
      xn, H_DIM, wqkvT, H_DIM, H_DIM, qkv, 3072,
      nullptr, nullptr, nullptr, nullptr, nullptr, nullptr, 0);
  rope_reorg_kernel<<<(BATCH * SEQ * NHEAD * 32) / 256, 256, 0, stream>>>(qkv, qro, kro, vtr);
  flash_attn_kernel<<<dim3(SEQ / 128, NHEAD, BATCH), 256, 0, stream>>>(qro, kro, vtr, attnb);
  gemm_bf16_kernel<<<dim3(H_DIM / 128, NTOK / 128), 256, 0, stream>>>(
      attnb, H_DIM, woT, H_DIM, H_DIM, x2, H_DIM,
      x, nullptr, nullptr, nullptr, nullptr, nullptr, 1);

  // 4) MoE path
  rmsnorm_kernel<<<NTOK, 256, 0, stream>>>(x2, g_ffn, xn2);
  router_kernel<<<NTOK / 8, 256, 0, stream>>>(xn2, w_router, etok, egate, cnt, probsum);
  for (int e = 0; e < NEXP; ++e) {
    // hidden = gelu(gather(xn2) @ w1[e])  -> bf16
    gemm_bf16_kernel<<<dim3(F_DIM / 128, NTOK / 128), 256, 0, stream>>>(
        xn2, H_DIM, w1t + (size_t)e * F_DIM * H_DIM, H_DIM, H_DIM,
        nullptr, F_DIM, nullptr, hidden,
        etok + e * NTOK, nullptr, nullptr, cnt + e, 2);
    // moe[token] += gate * (hidden @ w2[e])
    gemm_bf16_kernel<<<dim3(H_DIM / 128, NTOK / 128), 256, 0, stream>>>(
        hidden, F_DIM, w2t + (size_t)e * H_DIM * F_DIM, F_DIM, F_DIM,
        moe, H_DIM, nullptr, nullptr,
        nullptr, etok + e * NTOK, egate + e * NTOK, cnt + e, 3);
  }

  // 5) final residual + aux scalar
  add_kernel<<<(NTOK * H_DIM) / 1024, 256, 0, stream>>>(x2, moe, out);
  aux_kernel<<<1, 64, 0, stream>>>(cnt, probsum, out + (out_size - 1));
}
